// TransformerNMT_103079215703
// MI455X (gfx1250) — matrix operand order repaired and checked by reading, not compiled, not run on hardware
//
#include <hip/hip_runtime.h>
#include <hip/hip_bf16.h>

// ---------------- problem constants ----------------
#define Dm   512
#define Hh   8
#define HD   64         // head dim
#define Ll   6
#define DFFv 2048
#define Vv   32000
#define Bb   8
#define Ss   512        // src len == tgt len
#define Rr   (Bb*Ss)    // 4096 activation rows

typedef __attribute__((ext_vector_type(16))) __bf16 bf16x16;
typedef __attribute__((ext_vector_type(8)))  float  f32x8;

__device__ __forceinline__ __bf16 f2bf(float f) {
  unsigned u = __float_as_uint(f);
  unsigned r = u + 0x7FFFu + ((u >> 16) & 1u);   // RNE truncate
  unsigned short h = (unsigned short)(r >> 16);
  __bf16 b;
  __builtin_memcpy(&b, &h, sizeof(h));
  return b;
}

union FragU { bf16x16 v; uint4 q[2]; };

__device__ __forceinline__ bf16x16 load_frag(const __bf16* p0, const __bf16* p1) {
  FragU f;
  f.q[0] = *(const uint4*)p0;
  f.q[1] = *(const uint4*)p1;
  return f.v;
}

// 16-byte global -> LDS async copy (GLOBAL_LOAD_ASYNC_TO_LDS_B128, ASYNCcnt).
// Low 32 bits of a generic LDS pointer are the LDS byte address.
__device__ __forceinline__ void async_g2l_b128(const __bf16* g, __bf16* l) {
  unsigned loff = (unsigned)(unsigned long long)l;
  asm volatile("global_load_async_to_lds_b128 %0, %1, off"
               :: "v"(loff), "v"(g) : "memory");
}
__device__ __forceinline__ void wait_async0() {
  asm volatile("s_wait_asynccnt 0x0" ::: "memory");
}

// ---------------- batched bf16 GEMM: C = act(A*B^T + bias) ----------------
// A: [M,K] bf16 row-major (batch stride sA). B: [N,K] bf16 row-major (sB).
// K % 64 == 0, M % 128 == 0. N guarded. Output modes:
//   0: fp32 [M,N] (batch stride sC)      1: bf16 [M,N]
//   2: bf16 head-split   [B,H,S,hd]      3: bf16 head-split-T [B,H,hd,S]
//   4: bf16 merge-heads  [B,S,D] (batched over bh = blockIdx.z)
#define BM 128
#define BN 128
#define BK 64

__global__ void __launch_bounds__(256)
gemm_bf16_wmma(const __bf16* __restrict__ Ag, const __bf16* __restrict__ Bg,
               const float* __restrict__ bias, void* __restrict__ Cg,
               int M, int N, int K, int relu, int mode,
               long long sA, long long sB, long long sC)
{
  __shared__ __bf16 sAa[2 * BM * BK];   // 32 KB
  __shared__ __bf16 sBb[2 * BN * BK];   // 32 KB

  const __bf16* A = Ag + (long long)blockIdx.z * sA;
  const __bf16* B = Bg + (long long)blockIdx.z * sB;

  const int tid  = threadIdx.x;
  const int lane = tid & 31;
  const int wave = tid >> 5;
  const int wm   = wave & 3;            // 4 waves along M
  const int wn   = wave >> 2;           // 2 waves along N
  const int m0   = blockIdx.y * BM;
  const int n0   = blockIdx.x * BN;
  const int nlane = lane & 15;
  const int half  = lane >> 4;

  f32x8 acc[2][4];
#pragma unroll
  for (int a = 0; a < 2; ++a)
#pragma unroll
    for (int b = 0; b < 4; ++b) {
      f32x8 z = {0.f,0.f,0.f,0.f,0.f,0.f,0.f,0.f};
      acc[a][b] = z;
    }

  const int ntiles = K / BK;

  // ---- stage one k-tile (async b128, 4 A-chunks + 4 B-chunks per thread) ----
  auto stage = [&](int buf, int k0) {
    __bf16* la = sAa + buf * (BM * BK);
    __bf16* lb = sBb + buf * (BN * BK);
#pragma unroll
    for (int i = 0; i < 4; ++i) {
      int chunk = i * 256 + tid;        // 0..1023
      int r = chunk >> 3;               // tile row
      int c = (chunk & 7) * 8;          // bf16 col (16B chunk)
      async_g2l_b128(&A[(long long)(m0 + r) * K + k0 + c], &la[r * BK + c]);
      __bf16* lbp = &lb[r * BK + c];
      if (n0 + r < N) {
        async_g2l_b128(&B[(long long)(n0 + r) * K + k0 + c], lbp);
      } else {
        uint4 z; z.x = z.y = z.z = z.w = 0u;
        *(uint4*)lbp = z;
      }
    }
  };

  stage(0, 0);
  for (int t = 0; t < ntiles; ++t) {
    wait_async0();                      // this wave's fills of buf[t&1] done
    __syncthreads();                    // everyone's fills visible
    if (t + 1 < ntiles) stage((t + 1) & 1, (t + 1) * BK);

    const __bf16* la = sAa + (t & 1) * (BM * BK);
    const __bf16* lb = sBb + (t & 1) * (BN * BK);
#pragma unroll
    for (int kk = 0; kk < BK; kk += 32) {
      bf16x16 afr[2], bfr[4];
#pragma unroll
      for (int mi = 0; mi < 2; ++mi) {
        const __bf16* p = &la[(wm * 32 + mi * 16 + nlane) * BK + kk + half * 8];
        afr[mi] = load_frag(p, p + 16);   // K {0..7,16..23} / {8..15,24..31}
      }
#pragma unroll
      for (int ni = 0; ni < 4; ++ni) {
        const __bf16* p = &lb[(wn * 64 + ni * 16 + nlane) * BK + kk + half * 8];
        bfr[ni] = load_frag(p, p + 16);   // K {0..7,16..23} / {8..15,24..31}
      }
#pragma unroll
      for (int mi = 0; mi < 2; ++mi)
#pragma unroll
        for (int ni = 0; ni < 4; ++ni)
          acc[mi][ni] = __builtin_amdgcn_wmma_f32_16x16x32_bf16(
              false, afr[mi], false, bfr[ni], (short)0, acc[mi][ni], false, false);
    }
  }

  // ---- epilogue ----
  float*  Cf = (float*)Cg  + (mode == 0 ? (long long)blockIdx.z * sC : 0);
  __bf16* Cb = (__bf16*)Cg;
  const int mb = half * 8;
  const int z = blockIdx.z;
#pragma unroll
  for (int mi = 0; mi < 2; ++mi)
#pragma unroll
    for (int ni = 0; ni < 4; ++ni) {
      int col = n0 + wn * 64 + ni * 16 + nlane;
      if (col < N) {
        float bv = bias ? bias[col] : 0.f;
#pragma unroll
        for (int i = 0; i < 8; ++i) {
          int row = m0 + wm * 32 + mi * 16 + mb + i;
          if (row < M) {
            float v = acc[mi][ni][i] + bv;
            if (relu) v = fmaxf(v, 0.f);
            if (mode == 0) {
              Cf[(long long)row * N + col] = v;
            } else if (mode == 1) {
              Cb[(long long)row * N + col] = f2bf(v);
            } else if (mode == 2) {       // [R,D] -> [B,H,S,hd]
              int b = row >> 9, s = row & 511, h = col >> 6, d = col & 63;
              Cb[((((long long)b * Hh) + h) * Ss + s) * HD + d] = f2bf(v);
            } else if (mode == 3) {       // [R,D] -> [B,H,hd,S]
              int b = row >> 9, s = row & 511, h = col >> 6, d = col & 63;
              Cb[((((long long)b * Hh) + h) * HD + d) * Ss + s] = f2bf(v);
            } else {                      // batched bh: ctx [S,hd] -> [B,S,D]
              int b = z / Hh, h = z - b * Hh;
              Cb[(((long long)b * Ss + row) * Dm) + h * HD + col] = f2bf(v);
            }
          }
        }
      }
    }
}

// ---------------- weight convert + transpose: fp32 [K,N] -> bf16 [N,K] ----------------
__global__ void wconv_t(const float* __restrict__ in, __bf16* __restrict__ out,
                        int K, int N)
{
  const float* I = in  + (long long)blockIdx.z * K * N;
  __bf16*      O = out + (long long)blockIdx.z * K * N;
  long long idx = (long long)blockIdx.x * 256 + threadIdx.x;  // over N*K (out index)
  if (idx >= (long long)K * N) return;
  int k = (int)(idx % K);
  long long n = idx / K;
  O[idx] = f2bf(I[(long long)k * N + n]);
}

// ---------------- embedding + sinusoidal PE (fp32 + bf16 outputs) ----------------
__global__ void embed_pe(const int* __restrict__ tok, const float* __restrict__ emb,
                         float* __restrict__ out, __bf16* __restrict__ outb,
                         int S, float scale)
{
  long long idx = (long long)blockIdx.x * blockDim.x + threadIdx.x;  // over R*D
  int d = (int)(idx & (Dm - 1));
  long long rs = idx >> 9;
  int s = (int)(rs % S);
  int t = tok[rs];
  int i2 = d >> 1;
  float div = __expf(-(float)(2 * i2) * (9.210340371976184f / (float)Dm));
  float ang = (float)s * div;
  float pe = (d & 1) ? __cosf(ang) : __sinf(ang);
  float v = emb[(long long)t * Dm + d] * scale + pe;
  out[idx]  = v;
  outb[idx] = f2bf(v);
}

// ---------------- masked scaled softmax: fp32 scores in, bf16 probs out ----------------
__global__ void __launch_bounds__(256)
softmax_mask(float* __restrict__ scores, __bf16* __restrict__ probs,
             const int* __restrict__ tok, int mode, int Tq, int Tk, float scale)
{
  long long row = blockIdx.x;            // over B*H*Tq
  int q  = (int)(row % Tq);
  long long bh = row / Tq;
  long long b  = bh / Hh;
  float* s = scores + row * Tk;
  __bf16* p = probs + row * Tk;
  __shared__ float red[256];
  int tid = threadIdx.x;

  float vmax = -3.4e38f;
  for (int k = tid; k < Tk; k += 256) {
    float v = s[k] * scale;
    bool valid = (tok[b * Tk + k] != 0);
    if (mode == 1) valid = valid && (k <= q);
    v = valid ? v : -1e9f;
    s[k] = v;
    vmax = fmaxf(vmax, v);
  }
  red[tid] = vmax; __syncthreads();
  for (int off = 128; off > 0; off >>= 1) {
    if (tid < off) red[tid] = fmaxf(red[tid], red[tid + off]);
    __syncthreads();
  }
  vmax = red[0]; __syncthreads();

  float sum = 0.f;
  for (int k = tid; k < Tk; k += 256) {
    float e = __expf(s[k] - vmax);
    s[k] = e; sum += e;
  }
  red[tid] = sum; __syncthreads();
  for (int off = 128; off > 0; off >>= 1) {
    if (tid < off) red[tid] += red[tid + off];
    __syncthreads();
  }
  float inv = 1.f / red[0];
  for (int k = tid; k < Tk; k += 256) p[k] = f2bf(s[k] * inv);
}

// ---------------- fused residual + LayerNorm (fp32 + bf16 outputs) ----------------
__global__ void __launch_bounds__(256)
ln_residual(const float* __restrict__ x, const float* __restrict__ res,
            const float* __restrict__ g, const float* __restrict__ bb,
            float* __restrict__ out, __bf16* __restrict__ outb)
{
  long long row = blockIdx.x;
  const float* xr = x + row * Dm;
  const float* rr = res + row * Dm;
  float* orow = out + row * Dm;
  __bf16* obrow = outb + row * Dm;
  __shared__ float red[256];
  int tid = threadIdx.x;

  float s = 0.f;
  for (int d = tid; d < Dm; d += 256) s += xr[d] + rr[d];
  red[tid] = s; __syncthreads();
  for (int off = 128; off > 0; off >>= 1) {
    if (tid < off) red[tid] += red[tid + off];
    __syncthreads();
  }
  float mean = red[0] * (1.f / Dm); __syncthreads();

  float s2 = 0.f;
  for (int d = tid; d < Dm; d += 256) {
    float v = xr[d] + rr[d] - mean;
    s2 += v * v;
  }
  red[tid] = s2; __syncthreads();
  for (int off = 128; off > 0; off >>= 1) {
    if (tid < off) red[tid] += red[tid + off];
    __syncthreads();
  }
  float inv = rsqrtf(red[0] * (1.f / Dm) + 1e-5f);
  for (int d = tid; d < Dm; d += 256) {
    float v = (xr[d] + rr[d] - mean) * inv * g[d] + bb[d];
    orow[d]  = v;
    obrow[d] = f2bf(v);
  }
}

// ================= host orchestration =================
static inline void launch_gemm(const __bf16* A, const __bf16* B, const float* bias,
                               void* C, int M, int N, int K, int relu, int mode,
                               long long sA, long long sB, long long sC, int batch,
                               hipStream_t st)
{
  dim3 grid((N + BN - 1) / BN, (M + BM - 1) / BM, batch);
  gemm_bf16_wmma<<<grid, dim3(256), 0, st>>>(A, B, bias, C, M, N, K, relu, mode,
                                             sA, sB, sC);
}

struct Scratch {
  __bf16 *Qh, *Kh, *VhT, *CTXbf;    // ws, bf16
  float  *scoresF;                  // d_out scratch
  __bf16 *Pbf, *Hbf;                // d_out scratch
};

static void run_mha(const __bf16* q_in, const __bf16* kv_in,
                    const __bf16* Wt, const float* Bv,
                    int mode, const int* mtok, float* outF,
                    const Scratch& sc, hipStream_t st)
{
  const long long DD = (long long)Dm * Dm;
  launch_gemm(q_in,  Wt + 0 * DD, Bv + 0 * Dm, sc.Qh,  Rr, Dm, Dm, 0, 2, 0, 0, 0, 1, st);
  launch_gemm(kv_in, Wt + 1 * DD, Bv + 1 * Dm, sc.Kh,  Rr, Dm, Dm, 0, 2, 0, 0, 0, 1, st);
  launch_gemm(kv_in, Wt + 2 * DD, Bv + 2 * Dm, sc.VhT, Rr, Dm, Dm, 0, 3, 0, 0, 0, 1, st);
  // scores[bh] = Qh (S,hd) x Kh^T -> fp32 (S,S)
  launch_gemm(sc.Qh, sc.Kh, nullptr, sc.scoresF, Ss, Ss, HD, 0, 0,
              (long long)Ss * HD, (long long)Ss * HD, (long long)Ss * Ss, Bb * Hh, st);
  softmax_mask<<<Bb * Hh * Ss, 256, 0, st>>>(sc.scoresF, sc.Pbf, mtok, mode, Ss, Ss, 0.125f);
  // ctx[bh] = P (S,S) x VhT^T (=V) -> merged bf16 [B,S,D]
  launch_gemm(sc.Pbf, sc.VhT, nullptr, sc.CTXbf, Ss, HD, Ss, 0, 4,
              (long long)Ss * Ss, (long long)HD * Ss, 0, Bb * Hh, st);
  launch_gemm(sc.CTXbf, Wt + 3 * DD, Bv + 3 * Dm, outF, Rr, Dm, Dm, 0, 0, 0, 0, 0, 1, st);
}

extern "C" void kernel_launch(void* const* d_in, const int* in_sizes, int n_in,
                              void* d_out, int out_size, void* d_ws, size_t ws_size,
                              hipStream_t stream)
{
  const int*   src        = (const int*)  d_in[0];
  const int*   tgt        = (const int*)  d_in[1];
  const float* src_emb    = (const float*)d_in[2];
  const float* tgt_emb    = (const float*)d_in[3];
  const float* enc_attn_w = (const float*)d_in[4];
  const float* enc_attn_b = (const float*)d_in[5];
  const float* enc_ffn_w1 = (const float*)d_in[6];
  const float* enc_ffn_b1 = (const float*)d_in[7];
  const float* enc_ffn_w2 = (const float*)d_in[8];
  const float* enc_ffn_b2 = (const float*)d_in[9];
  const float* enc_ln_g   = (const float*)d_in[10];
  const float* enc_ln_b   = (const float*)d_in[11];
  const float* dec_self_w = (const float*)d_in[12];
  const float* dec_self_b = (const float*)d_in[13];
  const float* dec_cross_w= (const float*)d_in[14];
  const float* dec_cross_b= (const float*)d_in[15];
  const float* dec_ffn_w1 = (const float*)d_in[16];
  const float* dec_ffn_b1 = (const float*)d_in[17];
  const float* dec_ffn_w2 = (const float*)d_in[18];
  const float* dec_ffn_b2 = (const float*)d_in[19];
  const float* dec_ln_g   = (const float*)d_in[20];
  const float* dec_ln_b   = (const float*)d_in[21];
  const float* out_w      = (const float*)d_in[22];
  const float* out_b      = (const float*)d_in[23];

  const long long NB = (long long)Rr * Dm;     // 2,097,152 elements
  char* w = (char*)d_ws;
  long long off = 0;
  auto allocF = [&](long long n) { float*  p = (float*) (w + off); off += n * 4; return p; };
  auto allocB = [&](long long n) { __bf16* p = (__bf16*)(w + off); off += n * 2; return p; };

  float*  X    = allocF(NB);    // fp32 residual streams
  float*  ATT  = allocF(NB);
  float*  Y    = allocF(NB);
  __bf16* Xbf  = allocB(NB);
  __bf16* Ybf  = allocB(NB);
  Scratch sc;
  sc.Qh    = allocB(NB);
  sc.Kh    = allocB(NB);
  sc.VhT   = allocB(NB);
  sc.CTXbf = allocB(NB);
  // transposed bf16 weights
  const long long DD  = (long long)Dm * Dm;
  const long long FD  = (long long)Dm * DFFv;
  __bf16* WencAttnT = allocB(Ll * 4 * DD);
  __bf16* WdecSelfT = allocB(Ll * 4 * DD);
  __bf16* WdecCrossT= allocB(Ll * 4 * DD);
  __bf16* W1encT    = allocB(Ll * FD);
  __bf16* W2encT    = allocB(Ll * FD);
  __bf16* W1decT    = allocB(Ll * FD);
  __bf16* W2decT    = allocB(Ll * FD);
  __bf16* WoutT     = allocB((long long)Dm * Vv);

  // big scratch in d_out (final logits GEMM is the last writer)
  float* scratch = (float*)d_out;
  const long long SCN = (long long)Bb * Hh * Ss * Ss;       // 16.78M
  sc.scoresF = scratch;
  sc.Pbf = (__bf16*)(scratch + SCN);
  sc.Hbf = (__bf16*)(scratch + SCN + SCN / 2);

  // ---------- weight conversion (every call; deterministic) ----------
  {
    int gDD = (int)((DD + 255) / 256);
    wconv_t<<<dim3(gDD, 1, Ll * 4), 256, 0, stream>>>(enc_attn_w,  WencAttnT, Dm, Dm);
    wconv_t<<<dim3(gDD, 1, Ll * 4), 256, 0, stream>>>(dec_self_w,  WdecSelfT, Dm, Dm);
    wconv_t<<<dim3(gDD, 1, Ll * 4), 256, 0, stream>>>(dec_cross_w, WdecCrossT, Dm, Dm);
    int gFD = (int)((FD + 255) / 256);
    wconv_t<<<dim3(gFD, 1, Ll), 256, 0, stream>>>(enc_ffn_w1, W1encT, Dm, DFFv);
    wconv_t<<<dim3(gFD, 1, Ll), 256, 0, stream>>>(enc_ffn_w2, W2encT, DFFv, Dm);
    wconv_t<<<dim3(gFD, 1, Ll), 256, 0, stream>>>(dec_ffn_w1, W1decT, Dm, DFFv);
    wconv_t<<<dim3(gFD, 1, Ll), 256, 0, stream>>>(dec_ffn_w2, W2decT, DFFv, Dm);
    long long VN = (long long)Dm * Vv;
    wconv_t<<<dim3((int)((VN + 255) / 256), 1, 1), 256, 0, stream>>>(out_w, WoutT, Dm, Vv);
  }

  const float escale = 22.62741699796952f;  // sqrt(512)

  // ---------- encoder ----------
  embed_pe<<<(int)(NB / 256), 256, 0, stream>>>(src, src_emb, X, Xbf, Ss, escale);
  for (int i = 0; i < Ll; ++i) {
    run_mha(Xbf, Xbf, WencAttnT + (long long)i * 4 * DD,
            enc_attn_b + (long long)i * 4 * Dm, 0, src, ATT, sc, stream);
    ln_residual<<<Rr, 256, 0, stream>>>(ATT, X,
        enc_ln_g + ((long long)i * 2 + 0) * Dm, enc_ln_b + ((long long)i * 2 + 0) * Dm,
        X, Xbf);
    launch_gemm(Xbf, W1encT + (long long)i * FD, enc_ffn_b1 + (long long)i * DFFv,
                sc.Hbf, Rr, DFFv, Dm, 1, 1, 0, 0, 0, 1, stream);
    launch_gemm(sc.Hbf, W2encT + (long long)i * FD, enc_ffn_b2 + (long long)i * Dm,
                ATT, Rr, Dm, DFFv, 0, 0, 0, 0, 0, 1, stream);
    ln_residual<<<Rr, 256, 0, stream>>>(ATT, X,
        enc_ln_g + ((long long)i * 2 + 1) * Dm, enc_ln_b + ((long long)i * 2 + 1) * Dm,
        X, Xbf);
  }
  // Xbf is now the encoder memory (bf16)

  // ---------- decoder ----------
  embed_pe<<<(int)(NB / 256), 256, 0, stream>>>(tgt, tgt_emb, Y, Ybf, Ss, escale);
  for (int i = 0; i < Ll; ++i) {
    run_mha(Ybf, Ybf, WdecSelfT + (long long)i * 4 * DD,
            dec_self_b + (long long)i * 4 * Dm, 1, tgt, ATT, sc, stream);
    ln_residual<<<Rr, 256, 0, stream>>>(ATT, Y,
        dec_ln_g + ((long long)i * 3 + 0) * Dm, dec_ln_b + ((long long)i * 3 + 0) * Dm,
        Y, Ybf);
    run_mha(Ybf, Xbf, WdecCrossT + (long long)i * 4 * DD,
            dec_cross_b + (long long)i * 4 * Dm, 0, src, ATT, sc, stream);
    ln_residual<<<Rr, 256, 0, stream>>>(ATT, Y,
        dec_ln_g + ((long long)i * 3 + 1) * Dm, dec_ln_b + ((long long)i * 3 + 1) * Dm,
        Y, Ybf);
    launch_gemm(Ybf, W1decT + (long long)i * FD, dec_ffn_b1 + (long long)i * DFFv,
                sc.Hbf, Rr, DFFv, Dm, 1, 1, 0, 0, 0, 1, stream);
    launch_gemm(sc.Hbf, W2decT + (long long)i * FD, dec_ffn_b2 + (long long)i * Dm,
                ATT, Rr, Dm, DFFv, 0, 0, 0, 0, 0, 1, stream);
    ln_residual<<<Rr, 256, 0, stream>>>(ATT, Y,
        dec_ln_g + ((long long)i * 3 + 2) * Dm, dec_ln_b + ((long long)i * 3 + 2) * Dm,
        Y, Ybf);
  }

  // ---------- final projection: logits = Y @ out_w + out_b (overwrites scratch) ----------
  launch_gemm(Ybf, WoutT, out_b, d_out, Rr, Vv, Dm, 0, 0, 0, 0, 0, 1, stream);
  (void)in_sizes; (void)n_in; (void)out_size; (void)ws_size;
}
